// MSigLIP_50843822850318
// MI455X (gfx1250) — matrix operand order repaired and checked by reading, not compiled, not run on hardware
//
#include <hip/hip_runtime.h>
#include <hip/hip_bf16.h>

// ---------------------------------------------------------------------------
// SigLIP-base forward (image + text encoders) for gfx1250 (MI455X, wave32).
// All dense GEMMs go through v_wmma_f32_16x16x32_f16 (f16 in, f32 accum).
// Block tile 128x64, 4 waves; each wave owns a 32x64 strip (2x4 fragments,
// 8 WMMA per 32-K step). Next K-tile global loads are software-pipelined.
// ---------------------------------------------------------------------------

typedef _Float16 v16h __attribute__((ext_vector_type(16)));
typedef _Float16 v8h  __attribute__((ext_vector_type(8)));
typedef _Float16 v4h  __attribute__((ext_vector_type(4)));
typedef float    v8f  __attribute__((ext_vector_type(8)));
typedef float    v4f  __attribute__((ext_vector_type(4)));

#define H_   768
#define FF_  3072
#define NH_  12
#define B_   8
#define SIMG 196
#define STXT 64

static inline int cdiv(int a, int b) { return (a + b - 1) / b; }

__device__ __forceinline__ float gelu_tanh(float x) {
  return 0.5f * x * (1.0f + tanhf(0.7978845608028654f * (x + 0.044715f * x * x * x)));
}

// ---------------------------------------------------------------------------
// Generic batched WMMA GEMM:
//   C[z] = epilogue(alpha * A[z](M,K) x B[z](K,N))
// B layout: bnk==0 -> B[k*ldb+n] ("KN"), bnk==1 -> B[n*ldb+k] ("NK", X@W^T)
// Batch index z decomposes as (zb = z/nh, zh = z%nh) with separate strides.
// Epilogue: +bias[n], +res[(row % rmod)*ldres + n], optional tanh-GELU.
// All operand offsets / leading dims are multiples of 4 floats (16B aligned).
// ---------------------------------------------------------------------------
__global__ void __launch_bounds__(128) gemm_wmma_kernel(
    const float* __restrict__ A, long long sAb, long long sAh, int lda,
    const float* __restrict__ Bm, long long sBb, long long sBh, int ldb, int bnk,
    const float* __restrict__ bias,
    const float* __restrict__ res, int ldres, int rmod,
    float* __restrict__ C, long long sCb, long long sCh, int ldc,
    int M, int N, int K, int nh, float alpha, int do_gelu)
{
  __shared__ __align__(16) _Float16 sA[128][40]; // [m][k], row stride 80B
  __shared__ __align__(16) _Float16 sB[64][40];  // [n][k] (B transposed)

  const int tid  = threadIdx.x;
  const int wave = tid >> 5;
  const int lane = tid & 31;
  const int z    = blockIdx.z;
  const int zb   = z / nh, zh = z % nh;
  const size_t aoff = (size_t)(zb * sAb + zh * sAh);
  const size_t boff = (size_t)(zb * sBb + zh * sBh);
  const size_t coff = (size_t)(zb * sCb + zh * sCh);
  const int row0 = blockIdx.y * 128, col0 = blockIdx.x * 64;
  const int wm = wave * 32;            // wave's 32-row strip
  const int l16 = lane & 15;
  const int hiA = (lane >> 4) << 3;    // 0 or 8  (A: K-half select)
  const int hiB = (lane >> 4) << 4;    // 0 or 16 (unused; B fragments also use hiA)

  const bool mFull = (row0 + 128 <= M);
  const bool nFull = (col0 + 64 <= N);

  float regA[32];   // staged A tile slice (128x32 / 128 thr = 32 f32)
  float regB[16];   // staged B tile slice (64x32 / 128 thr = 16 f32)

  // Unified element ordering: j -> f = tid + (j/4)*128, e = f*4 + (j%4).
  auto loadA = [&](int k0) {
    if (mFull && (k0 + 32 <= K)) {
      #pragma unroll
      for (int g = 0; g < 8; ++g) {
        int f = tid + g * 128;
        int r = f >> 3, c = (f & 7) * 4;
        v4f t = *reinterpret_cast<const v4f*>(&A[aoff + (size_t)(row0 + r) * lda + k0 + c]);
        #pragma unroll
        for (int i = 0; i < 4; ++i) regA[g * 4 + i] = t[i];
      }
    } else {
      #pragma unroll
      for (int j = 0; j < 32; ++j) {
        int f = tid + (j >> 2) * 128;
        int e = f * 4 + (j & 3);
        int r = e >> 5, c = e & 31;
        int gr = row0 + r, gk = k0 + c;
        int cr = gr < M ? gr : (M - 1);
        int ck = gk < K ? gk : (K - 1);
        float v = A[aoff + (size_t)cr * lda + ck];
        regA[j] = (gr < M && gk < K) ? v : 0.0f;
      }
    }
  };
  auto storeA = [&]() {
    #pragma unroll
    for (int j = 0; j < 32; ++j) {
      int f = tid + (j >> 2) * 128;
      int e = f * 4 + (j & 3);
      sA[e >> 5][e & 31] = (_Float16)regA[j];
    }
  };
  auto loadB = [&](int k0) {
    bool fast = nFull && (k0 + 32 <= K);
    if (fast && bnk) {
      #pragma unroll
      for (int g = 0; g < 4; ++g) {
        int f = tid + g * 128;
        int n = f >> 3, c = (f & 7) * 4;
        v4f t = *reinterpret_cast<const v4f*>(&Bm[boff + (size_t)(col0 + n) * ldb + k0 + c]);
        #pragma unroll
        for (int i = 0; i < 4; ++i) regB[g * 4 + i] = t[i];
      }
    } else if (fast) {
      #pragma unroll
      for (int g = 0; g < 4; ++g) {
        int f = tid + g * 128;
        int k = f >> 4, n4 = (f & 15) * 4;
        v4f t = *reinterpret_cast<const v4f*>(&Bm[boff + (size_t)(k0 + k) * ldb + col0 + n4]);
        #pragma unroll
        for (int i = 0; i < 4; ++i) regB[g * 4 + i] = t[i];
      }
    } else {
      #pragma unroll
      for (int j = 0; j < 16; ++j) {
        int e = tid + j * 128;
        int n = e >> 5, c = e & 31;
        int gn = col0 + n, gk = k0 + c;
        int cn = gn < N ? gn : (N - 1);
        int ck = gk < K ? gk : (K - 1);
        float v = bnk ? Bm[boff + (size_t)cn * ldb + ck]
                      : Bm[boff + (size_t)ck * ldb + cn];
        regB[j] = (gn < N && gk < K) ? v : 0.0f;
      }
    }
  };
  auto storeB = [&](int k0) {
    bool fast = nFull && (k0 + 32 <= K);
    if (fast && bnk) {
      #pragma unroll
      for (int j = 0; j < 16; ++j) {
        int f = tid + (j >> 2) * 128;
        int e = f * 4 + (j & 3);
        sB[e >> 5][e & 31] = (_Float16)regB[j];
      }
    } else if (fast) {
      #pragma unroll
      for (int j = 0; j < 16; ++j) {
        int f = tid + (j >> 2) * 128;
        int k = f >> 4, n4 = (f & 15) * 4;
        sB[n4 + (j & 3)][k] = (_Float16)regB[j];
      }
    } else {
      #pragma unroll
      for (int j = 0; j < 16; ++j) {
        int e = tid + j * 128;
        sB[e >> 5][e & 31] = (_Float16)regB[j];
      }
    }
  };

  v8f acc[2][4] = {};

  loadA(0);
  loadB(0);
  for (int k0 = 0; k0 < K; k0 += 32) {
    storeA();
    storeB(k0);
    __syncthreads();

    // software pipeline: issue next tile's global loads before compute
    int k1 = k0 + 32;
    if (k1 < K) { loadA(k1); loadB(k1); }

    v16h afrag[2], bfrag[4];
    #pragma unroll
    for (int mi = 0; mi < 2; ++mi) {
      int m = wm + mi * 16 + l16;
      v8h a0 = *reinterpret_cast<const v8h*>(&sA[m][hiA]);
      v8h a1 = *reinterpret_cast<const v8h*>(&sA[m][16 + hiA]);
      #pragma unroll
      for (int i = 0; i < 8; ++i) { afrag[mi][i] = a0[i]; afrag[mi][i + 8] = a1[i]; }
    }
    #pragma unroll
    for (int ni = 0; ni < 4; ++ni) {
      int n = ni * 16 + l16;
      v8h b0 = *reinterpret_cast<const v8h*>(&sB[n][hiA]);
      v8h b1 = *reinterpret_cast<const v8h*>(&sB[n][16 + hiA]);
      #pragma unroll
      for (int i = 0; i < 8; ++i) { bfrag[ni][i] = b0[i]; bfrag[ni][i + 8] = b1[i]; }
    }
    #pragma unroll
    for (int mi = 0; mi < 2; ++mi)
      #pragma unroll
      for (int ni = 0; ni < 4; ++ni)
        acc[mi][ni] = __builtin_amdgcn_wmma_f32_16x16x32_f16(
            false, afrag[mi], false, bfrag[ni], (short)0, acc[mi][ni], false, false);
    __syncthreads();
  }

  // ---- epilogue / store (D layout: lane%16 = N, vgpr + 8*(lane/16) = M) ----
  const int mBase = (lane >> 4) << 3;
  #pragma unroll
  for (int mi = 0; mi < 2; ++mi) {
    #pragma unroll
    for (int ni = 0; ni < 4; ++ni) {
      #pragma unroll
      for (int v = 0; v < 8; ++v) {
        int row = row0 + wm + mi * 16 + mBase + v;
        int col = col0 + ni * 16 + l16;
        if (row < M && col < N) {
          float val = acc[mi][ni][v] * alpha;
          if (bias) val += bias[col];
          if (res) {
            int rr = (rmod > 0) ? (row % rmod) : row;
            val += res[(size_t)rr * ldres + col];
          }
          if (do_gelu) val = gelu_tanh(val);
          C[coff + (size_t)row * ldc + col] = val;
        }
      }
    }
  }
}

// ---------------------------------------------------------------------------
// LayerNorm over H=768 per row (one block per row, 256 threads)
// ---------------------------------------------------------------------------
__global__ void ln_kernel(const float* __restrict__ x, const float* __restrict__ s,
                          const float* __restrict__ b, float* __restrict__ y, int M)
{
  int row = blockIdx.x;
  if (row >= M) return;
  const float* xr = x + (size_t)row * H_;
  __shared__ float red[256];
  int tid = threadIdx.x;

  float lsum = 0.0f;
  for (int i = tid; i < H_; i += 256) lsum += xr[i];
  red[tid] = lsum; __syncthreads();
  for (int o = 128; o > 0; o >>= 1) { if (tid < o) red[tid] += red[tid + o]; __syncthreads(); }
  float mean = red[0] * (1.0f / H_);
  __syncthreads();

  float lv = 0.0f;
  for (int i = tid; i < H_; i += 256) { float d = xr[i] - mean; lv += d * d; }
  red[tid] = lv; __syncthreads();
  for (int o = 128; o > 0; o >>= 1) { if (tid < o) red[tid] += red[tid + o]; __syncthreads(); }
  float rstd = rsqrtf(red[0] * (1.0f / H_) + 1e-6f);

  for (int i = tid; i < H_; i += 256)
    y[(size_t)row * H_ + i] = (xr[i] - mean) * rstd * s[i] + b[i];
}

// ---------------------------------------------------------------------------
// Row softmax (len <= 256), one block per row
// ---------------------------------------------------------------------------
__global__ void softmax_kernel(float* __restrict__ sc, int len)
{
  size_t row = blockIdx.x;
  float* p = sc + row * (size_t)len;
  __shared__ float red[256];
  int tid = threadIdx.x;

  float v = (tid < len) ? p[tid] : -1e30f;
  red[tid] = v; __syncthreads();
  for (int o = 128; o > 0; o >>= 1) { if (tid < o) red[tid] = fmaxf(red[tid], red[tid + o]); __syncthreads(); }
  float m = red[0]; __syncthreads();

  float e = (tid < len) ? expf(v - m) : 0.0f;
  red[tid] = e; __syncthreads();
  for (int o = 128; o > 0; o >>= 1) { if (tid < o) red[tid] += red[tid + o]; __syncthreads(); }
  float s = red[0];
  if (tid < len) p[tid] = e / s;
}

// ---------------------------------------------------------------------------
// Small utility kernels
// ---------------------------------------------------------------------------
__global__ void im2col_kernel(const float* __restrict__ img, float* __restrict__ Xp)
{
  size_t idx = (size_t)blockIdx.x * 256 + threadIdx.x;
  const size_t total = (size_t)B_ * SIMG * H_;
  if (idx >= total) return;
  int col = (int)(idx % H_);
  size_t row = idx / H_;
  int p = (int)(row % SIMG);
  int b = (int)(row / SIMG);
  int py = p / 14, px = p % 14;
  int c = col / 256, rem = col % 256, ky = rem / 16, kx = rem % 16;
  Xp[idx] = img[(((size_t)b * 3 + c) * 224 + (py * 16 + ky)) * 224 + (px * 16 + kx)];
}

__global__ void embed_text_kernel(const int* __restrict__ ids, const float* __restrict__ tok,
                                  const float* __restrict__ pos, float* __restrict__ x)
{
  size_t idx = (size_t)blockIdx.x * 256 + threadIdx.x;
  const size_t total = (size_t)B_ * STXT * H_;
  if (idx >= total) return;
  int k = (int)(idx % H_);
  size_t row = idx / H_;
  int t = (int)(row % STXT);
  int b = (int)(row / STXT);
  int id = ids[b * STXT + t];
  id = id < 0 ? 0 : (id > 31999 ? 31999 : id);
  x[idx] = tok[(size_t)id * H_ + k] + pos[(size_t)t * H_ + k];
}

__global__ void avg_heads_kernel(const float* __restrict__ pr, float* __restrict__ am, int S)
{
  size_t idx = (size_t)blockIdx.x * 256 + threadIdx.x;
  const size_t total = (size_t)B_ * S * S;
  if (idx >= total) return;
  size_t b = idx / ((size_t)S * S);
  size_t r = idx % ((size_t)S * S);
  float a = 0.0f;
  for (int h = 0; h < NH_; ++h) a += pr[((size_t)b * NH_ + h) * S * S + r];
  am[idx] = a * (1.0f / NH_);
}

__global__ void pi_kernel(const float* __restrict__ am, float* __restrict__ pi, int S)
{
  int b = blockIdx.x;
  int j = threadIdx.x;
  __shared__ float red[256];
  float pr = 0.0f;
  if (j < S) {
    float a = 0.0f;
    for (int i = 0; i < S; ++i) a += am[((size_t)b * S + i) * S + j];
    pr = a * (1.0f / S);
  }
  red[j] = pr; __syncthreads();
  for (int o = 128; o > 0; o >>= 1) { if (j < o) red[j] += red[j + o]; __syncthreads(); }
  float tot = red[0];
  if (j < S) pi[(size_t)b * S + j] = pr / (tot + 1e-6f);
}

__global__ void fake_atten_kernel(const float* __restrict__ am, const float* __restrict__ pi,
                                  float* __restrict__ out, int S, int uniform)
{
  int SP = S + 1;
  size_t idx = (size_t)blockIdx.x * 256 + threadIdx.x;
  const size_t total = (size_t)B_ * SP * SP;
  if (idx >= total) return;
  size_t b = idx / ((size_t)SP * SP);
  size_t r = idx % ((size_t)SP * SP);
  int i = (int)(r / SP), j = (int)(r % SP);
  float v;
  if (i == 0 && j == 0)      v = 1.0f;
  else if (i == 0)           v = uniform ? (1.0f / S) : pi[b * S + (j - 1)];
  else if (j == 0)           v = 1.0f / S;
  else                       v = am[(b * S + (i - 1)) * S + (j - 1)];
  out[idx] = v;
}

__global__ void scatter_feats_kernel(const float* __restrict__ src, float* __restrict__ dst, int S)
{
  size_t idx = (size_t)blockIdx.x * 256 + threadIdx.x;
  const size_t total = (size_t)B_ * S * H_;
  if (idx >= total) return;
  size_t b = idx / ((size_t)S * H_);
  size_t r = idx % ((size_t)S * H_);
  int s2 = (int)(r / H_), k = (int)(r % H_);
  dst[((size_t)b * (S + 1) + 1 + s2) * H_ + k] = src[idx];
}

__global__ void qvec_kernel(const float* __restrict__ probe, const float* __restrict__ w,
                            const float* __restrict__ bias, float* __restrict__ qv)
{
  for (int n = threadIdx.x; n < H_; n += blockDim.x) {
    const float* wr = w + (size_t)n * H_;
    float a = 0.0f;
    for (int k = 0; k < H_; ++k) a += probe[k] * wr[k];
    qv[n] = a + bias[n];
  }
}

// Attention-pooling head: single query per (b, head); block per (b,h), 256 thr.
__global__ void pool_attn_kernel(const float* __restrict__ qv, const float* __restrict__ Kh,
                                 const float* __restrict__ Vh, float* __restrict__ o, int S)
{
  int z = blockIdx.x;
  int b = z / NH_, h = z % NH_;
  __shared__ float prob[256];
  __shared__ float red[256];
  int tid = threadIdx.x;

  float logit = -1e30f;
  if (tid < S) {
    const float* kr = Kh + ((size_t)(b * S + tid)) * H_ + h * 64;
    float acc = 0.0f;
    for (int d = 0; d < 64; ++d) acc += qv[h * 64 + d] * kr[d];
    logit = acc * 0.125f;
  }
  red[tid] = logit; __syncthreads();
  for (int ofs = 128; ofs > 0; ofs >>= 1) { if (tid < ofs) red[tid] = fmaxf(red[tid], red[tid + ofs]); __syncthreads(); }
  float m = red[0]; __syncthreads();

  float e = (tid < S) ? expf(logit - m) : 0.0f;
  red[tid] = e; __syncthreads();
  for (int ofs = 128; ofs > 0; ofs >>= 1) { if (tid < ofs) red[tid] += red[tid + ofs]; __syncthreads(); }
  float ssum = red[0];
  prob[tid] = e / ssum;
  __syncthreads();

  if (tid < 64) {
    float acc = 0.0f;
    for (int s2 = 0; s2 < S; ++s2)
      acc += prob[s2] * Vh[((size_t)(b * S + s2)) * H_ + h * 64 + tid];
    o[(size_t)b * H_ + h * 64 + tid] = acc;
  }
}

// ---------------------------------------------------------------------------
// Host-side orchestration
// ---------------------------------------------------------------------------
static inline void gemm(hipStream_t st,
    const float* A, long long sAb, long long sAh, int lda,
    const float* Bm, long long sBb, long long sBh, int ldb, int bnk,
    const float* bias, const float* res, int ldres, int rmod,
    float* C, long long sCb, long long sCh, int ldc,
    int M, int N, int K, int batch, int nh, float alpha, int do_gelu)
{
  dim3 grid(cdiv(N, 64), cdiv(M, 128), batch);
  gemm_wmma_kernel<<<grid, 128, 0, st>>>(A, sAb, sAh, lda, Bm, sBb, sBh, ldb, bnk,
                                         bias, res, ldres, rmod, C, sCb, sCh, ldc,
                                         M, N, K, nh, alpha, do_gelu);
}

struct EncW {
  const float *ln1_s, *ln1_b, *wq, *bq, *wk, *bk, *wv, *bv, *wo, *bo;
  const float *ln2_s, *ln2_b, *fc1_w, *fc1_b, *fc2_w, *fc2_b;
};

static void run_encoder(hipStream_t st, float* hA, float* hB, float* Xb,
                        float* Qb, float* Kb, float* Vb, float* Ob,
                        float* FFb, float* Sc, const EncW& w, int S)
{
  const int M = B_ * S;
  for (int li = 0; li < 12; ++li) {
    const size_t oH  = (size_t)li * H_;
    const size_t oHH = (size_t)li * H_ * H_;
    const size_t oHF = (size_t)li * H_ * FF_;
    const size_t oF  = (size_t)li * FF_;

    ln_kernel<<<M, 256, 0, st>>>(hA, w.ln1_s + oH, w.ln1_b + oH, Xb, M);
    gemm(st, Xb, 0, 0, H_, w.wq + oHH, 0, 0, H_, 0, w.bq + oH, nullptr, 0, 0,
         Qb, 0, 0, H_, M, H_, H_, 1, 1, 1.0f, 0);
    gemm(st, Xb, 0, 0, H_, w.wk + oHH, 0, 0, H_, 0, w.bk + oH, nullptr, 0, 0,
         Kb, 0, 0, H_, M, H_, H_, 1, 1, 1.0f, 0);
    gemm(st, Xb, 0, 0, H_, w.wv + oHH, 0, 0, H_, 0, w.bv + oH, nullptr, 0, 0,
         Vb, 0, 0, H_, M, H_, H_, 1, 1, 1.0f, 0);

    // scores[b,h] = (Q_bh @ K_bh^T) * SCALE
    gemm(st, Qb, (long long)S * H_, 64, H_,
             Kb, (long long)S * H_, 64, H_, 1,
             nullptr, nullptr, 0, 0,
             Sc, (long long)NH_ * S * S, (long long)S * S, S,
             S, S, 64, B_ * NH_, NH_, 0.125f, 0);
    softmax_kernel<<<B_ * NH_ * S, 256, 0, st>>>(Sc, S);
    // O[b,h] = A_bh @ V_bh
    gemm(st, Sc, (long long)NH_ * S * S, (long long)S * S, S,
             Vb, (long long)S * H_, 64, H_, 0,
             nullptr, nullptr, 0, 0,
             Ob, (long long)S * H_, 64, H_,
             S, 64, S, B_ * NH_, NH_, 1.0f, 0);

    // hB = hA + O @ wo + bo
    gemm(st, Ob, 0, 0, H_, w.wo + oHH, 0, 0, H_, 0, w.bo + oH, hA, H_, 0,
         hB, 0, 0, H_, M, H_, H_, 1, 1, 1.0f, 0);

    ln_kernel<<<M, 256, 0, st>>>(hB, w.ln2_s + oH, w.ln2_b + oH, Xb, M);
    gemm(st, Xb, 0, 0, H_, w.fc1_w + oHF, 0, 0, FF_, 0, w.fc1_b + oF, nullptr, 0, 0,
         FFb, 0, 0, FF_, M, FF_, H_, 1, 1, 1.0f, 1);
    gemm(st, FFb, 0, 0, FF_, w.fc2_w + oHF, 0, 0, H_, 0, w.fc2_b + oH, hB, H_, 0,
         hA, 0, 0, H_, M, H_, FF_, 1, 1, 1.0f, 0);
  }
}

static EncW make_encw(void* const* d, int base)
{
  EncW w;
  w.ln1_s = (const float*)d[base + 0];  w.ln1_b = (const float*)d[base + 1];
  w.wq    = (const float*)d[base + 2];  w.bq    = (const float*)d[base + 3];
  w.wk    = (const float*)d[base + 4];  w.bk    = (const float*)d[base + 5];
  w.wv    = (const float*)d[base + 6];  w.bv    = (const float*)d[base + 7];
  w.wo    = (const float*)d[base + 8];  w.bo    = (const float*)d[base + 9];
  w.ln2_s = (const float*)d[base + 10]; w.ln2_b = (const float*)d[base + 11];
  w.fc1_w = (const float*)d[base + 12]; w.fc1_b = (const float*)d[base + 13];
  w.fc2_w = (const float*)d[base + 14]; w.fc2_b = (const float*)d[base + 15];
  return w;
}

extern "C" void kernel_launch(void* const* d_in, const int* in_sizes, int n_in,
                              void* d_out, int out_size, void* d_ws, size_t ws_size,
                              hipStream_t stream)
{
  (void)in_sizes; (void)n_in; (void)out_size; (void)ws_size;
  hipStream_t st = stream;

  // ---- inputs (setup_inputs order, dicts flattened in insertion order) ----
  const float* image   = (const float*)d_in[0];
  const int*   text    = (const int*)d_in[1];
  const float* patch_w = (const float*)d_in[2];
  const float* patch_b = (const float*)d_in[3];
  const float* vpos    = (const float*)d_in[4];
  EncW vw = make_encw(d_in, 5);
  const float* post_s    = (const float*)d_in[21];
  const float* post_b    = (const float*)d_in[22];
  const float* probe     = (const float*)d_in[23];
  const float* head_in_w = (const float*)d_in[24];
  const float* head_in_b = (const float*)d_in[25];
  const float* head_out_w= (const float*)d_in[26];
  const float* head_out_b= (const float*)d_in[27];
  const float* head_ln_s = (const float*)d_in[28];
  const float* head_ln_b = (const float*)d_in[29];
  const float* head_fc1_w= (const float*)d_in[30];
  const float* head_fc1_b= (const float*)d_in[31];
  const float* head_fc2_w= (const float*)d_in[32];
  const float* head_fc2_b= (const float*)d_in[33];
  const float* tok       = (const float*)d_in[34];
  const float* tpos      = (const float*)d_in[35];
  EncW tw = make_encw(d_in, 36);
  const float* fin_s  = (const float*)d_in[52];
  const float* fin_b  = (const float*)d_in[53];
  const float* thead_w= (const float*)d_in[54];
  const float* thead_b= (const float*)d_in[55];

  // ---- outputs (concat order: image_feats, atten_i, text_feats, atten_t) ----
  float* out = (float*)d_out;
  float* o_img_feats = out;                                   // 8*197*768
  float* o_atten_i   = out + (size_t)B_ * 197 * H_;           // 8*197*197
  float* o_txt_feats = o_atten_i + (size_t)B_ * 197 * 197;    // 8*65*768
  float* o_atten_t   = o_txt_feats + (size_t)B_ * 65 * H_;    // 8*65*65

  // ---- workspace layout (floats), ~69 MB total ----
  float* w = (float*)d_ws;
  const size_t BUF = (size_t)B_ * SIMG * H_;                  // 1,204,224
  float* hA  = w;
  float* hB  = w + 1 * BUF;
  float* Xb  = w + 2 * BUF;
  float* Qb  = w + 3 * BUF;
  float* Kb  = w + 4 * BUF;
  float* Vb  = w + 5 * BUF;
  float* Ob  = w + 6 * BUF;
  float* FFb = w + 7 * BUF;                                   // 8*196*3072
  float* Sc  = FFb + (size_t)B_ * SIMG * FF_;                 // 8*12*196*196
  float* am  = Sc + (size_t)B_ * NH_ * SIMG * SIMG;           // 8*196*196
  float* pi  = am + (size_t)B_ * SIMG * SIMG;                 // 8*196
  float* qv  = pi + (size_t)B_ * SIMG;                        // 768
  float* opool = qv + H_;                                     // 8*768
  float* xb2 = opool + (size_t)B_ * H_;                       // 8*768
  float* yb2 = xb2 + (size_t)B_ * H_;                         // 8*768
  float* tb2 = yb2 + (size_t)B_ * H_;                         // 8*3072

  // =========================== IMAGE ENCODER ===========================
  {
    const int M = B_ * SIMG;
    // patch-embed: im2col -> GEMM(X @ patch_w^T + patch_b + pos)
    im2col_kernel<<<cdiv((int)(B_ * SIMG * H_), 256), 256, 0, st>>>(image, Ob);
    gemm(st, Ob, 0, 0, H_, patch_w, 0, 0, H_, 1, patch_b,
         vpos, H_, SIMG, hA, 0, 0, H_, M, H_, H_, 1, 1, 1.0f, 0);

    run_encoder(st, hA, hB, Xb, Qb, Kb, Vb, Ob, FFb, Sc, vw, SIMG);

    // post-LN, feats rows 1..196
    ln_kernel<<<M, 256, 0, st>>>(hA, post_s, post_b, Xb, M);
    scatter_feats_kernel<<<cdiv((int)(B_ * SIMG * H_), 256), 256, 0, st>>>(Xb, o_img_feats, SIMG);

    // attention-pooling head
    qvec_kernel<<<1, 256, 0, st>>>(probe, head_in_w, head_in_b, qv);
    gemm(st, Xb, 0, 0, H_, head_in_w + (size_t)H_ * H_, 0, 0, H_, 1,
         head_in_b + H_, nullptr, 0, 0, Kb, 0, 0, H_, M, H_, H_, 1, 1, 1.0f, 0);
    gemm(st, Xb, 0, 0, H_, head_in_w + (size_t)2 * H_ * H_, 0, 0, H_, 1,
         head_in_b + 2 * H_, nullptr, 0, 0, Vb, 0, 0, H_, M, H_, H_, 1, 1, 1.0f, 0);
    pool_attn_kernel<<<B_ * NH_, 256, 0, st>>>(qv, Kb, Vb, opool, SIMG);
    gemm(st, opool, 0, 0, H_, head_out_w, 0, 0, H_, 1, head_out_b, nullptr, 0, 0,
         xb2, 0, 0, H_, B_, H_, H_, 1, 1, 1.0f, 0);
    ln_kernel<<<B_, 256, 0, st>>>(xb2, head_ln_s, head_ln_b, yb2, B_);
    gemm(st, yb2, 0, 0, H_, head_fc1_w, 0, 0, FF_, 0, head_fc1_b, nullptr, 0, 0,
         tb2, 0, 0, FF_, B_, FF_, H_, 1, 1, 1.0f, 1);
    // pooled -> row 0 of each image's feats (ldc = 197*768)
    gemm(st, tb2, 0, 0, FF_, head_fc2_w, 0, 0, H_, 0, head_fc2_b, xb2, H_, 0,
         o_img_feats, 0, 0, 197 * H_, B_, H_, FF_, 1, 1, 1.0f, 0);

    // fake attention map from last-layer probs (still in Sc)
    avg_heads_kernel<<<cdiv((int)(B_ * SIMG * SIMG), 256), 256, 0, st>>>(Sc, am, SIMG);
    pi_kernel<<<B_, 256, 0, st>>>(am, pi, SIMG);
    fake_atten_kernel<<<cdiv((int)(B_ * 197 * 197), 256), 256, 0, st>>>(am, pi, o_atten_i, SIMG, 0);
  }

  // =========================== TEXT ENCODER ============================
  {
    const int M = B_ * STXT;
    embed_text_kernel<<<cdiv((int)(B_ * STXT * H_), 256), 256, 0, st>>>(text, tok, tpos, hA);

    run_encoder(st, hA, hB, Xb, Qb, Kb, Vb, Ob, FFb, Sc, tw, STXT);

    avg_heads_kernel<<<cdiv((int)(B_ * STXT * STXT), 256), 256, 0, st>>>(Sc, am, STXT);
    fake_atten_kernel<<<cdiv((int)(B_ * 65 * 65), 256), 256, 0, st>>>(am, nullptr, o_atten_t, STXT, 1);

    ln_kernel<<<M, 256, 0, st>>>(hA, fin_s, fin_b, Xb, M);
    scatter_feats_kernel<<<cdiv((int)(B_ * STXT * H_), 256), 256, 0, st>>>(Xb, o_txt_feats, STXT);
    // pooled = h_ln[:, -1, :] @ head_w^T + head_b -> row 0 (ldc = 65*768)
    gemm(st, Xb + (size_t)(STXT - 1) * H_, 0, 0, STXT * H_,
         thead_w, 0, 0, H_, 1, thead_b, nullptr, 0, 0,
         o_txt_feats, 0, 0, 65 * H_, B_, H_, H_, 1, 1, 1.0f, 0);
  }
}
